// PyTorchAttention_65996467470660
// MI455X (gfx1250) — compile-verified, matrix operand order checked by reading, not run on hardware
//
#include <hip/hip_runtime.h>

typedef __attribute__((ext_vector_type(16))) __bf16 v16bf;
typedef __attribute__((ext_vector_type(8)))  __bf16 v8bf;
typedef __attribute__((ext_vector_type(4)))  __bf16 v4bf;
typedef __attribute__((ext_vector_type(8)))  float  v8f;

#define D_MODEL 256
#define SEQ_LEN 4096
#define BK      64      // keys per block iteration
#define WAVES   8       // waves per workgroup
#define BQ      16      // queries per wave (WMMA M tile)
// 1/sqrt(256) * log2(e): scores land directly in base-2 domain -> exp2f == v_exp_f32
#define QSCALE  (0.0625f * 1.44269504088896340736f)

#define SMEM_BYTES ((2 * BK * D_MODEL + WAVES * BQ * BK) * (int)sizeof(__bf16))  // 80 KB

// Build a 16x32 bf16 fragment (per-lane: two contiguous 8-element chunks).
static __device__ __forceinline__ v16bf frag_from_lds(const __bf16* p0, const __bf16* p1) {
  v8bf lo = *(const v8bf*)p0;
  v8bf hi = *(const v8bf*)p1;
  return __builtin_shufflevector(lo, hi, 0,1,2,3,4,5,6,7,8,9,10,11,12,13,14,15);
}

static __device__ __forceinline__ v16bf frag_from_global_scaled(const float* p0, const float* p1) {
  v16bf r;
#pragma unroll
  for (int i = 0; i < 8; ++i) {
    r[i]     = (__bf16)(p0[i] * QSCALE);
    r[i + 8] = (__bf16)(p1[i] * QSCALE);
  }
  return r;
}

__global__ __launch_bounds__(256, 1)
void flash_attn_bf16_wmma(const float* __restrict__ Q, const float* __restrict__ K,
                          const float* __restrict__ V, float* __restrict__ Out,
                          int S, int B) {
  extern __shared__ __bf16 smem[];
  __bf16* sK  = smem;                         // [BK][D_MODEL] row-major       (32 KB)
  __bf16* sVt = smem + BK * D_MODEL;          // [D_MODEL][BK]  (transposed V) (32 KB)

  const int tid  = threadIdx.x;
  const int wave = tid >> 5;
  const int lane = tid & 31;
  const int b    = blockIdx.y;
  const int q0   = blockIdx.x * (WAVES * BQ) + wave * BQ;

  __bf16* sPw = smem + 2 * BK * D_MODEL + wave * (BQ * BK);  // per-wave [BQ][BK] (2 KB)

  const int r    = lane & 15;            // fragment row within 16
  const int koff = (lane < 16) ? 0 : 8;  // per-lane-half K-chunk offset (ISA A/B layout)

  // ---- Q tile (16 x 256) -> 8 bf16 A-fragments, pre-scaled by 1/sqrt(d)*log2(e) ----
  const float* Qrow = Q + ((size_t)(b * S + q0 + r)) * D_MODEL;
  v16bf qf[8];
#pragma unroll
  for (int f = 0; f < 8; ++f)
    qf[f] = frag_from_global_scaled(Qrow + f * 32 + koff, Qrow + f * 32 + 16 + koff);

  // ---- Accumulators / online-softmax state ----
  v8f o[16];
#pragma unroll
  for (int n = 0; n < 16; ++n) o[n] = (v8f)(0.0f);
  float m[8], l[8];
#pragma unroll
  for (int v = 0; v < 8; ++v) { m[v] = -INFINITY; l[v] = 0.0f; }

  for (int kb = 0; kb < S; kb += BK) {
    __syncthreads();  // previous tile fully consumed by all waves

    // ---- Cooperative tile load: 64 keys x 256 dims, fp32 -> bf16 ----
#pragma unroll
    for (int t = 0; t < 16; ++t) {
      const int idx = tid + t * 256;
      const int row = idx >> 6;        // 0..63 key within block
      const int c4  = idx & 63;        // float4 column
      const size_t gbase = ((size_t)(b * S + kb + row)) * D_MODEL + c4 * 4;
      const float4 kv = *(const float4*)(K + gbase);
      v4bf kp = { (__bf16)kv.x, (__bf16)kv.y, (__bf16)kv.z, (__bf16)kv.w };
      *(v4bf*)&sK[row * D_MODEL + c4 * 4] = kp;   // one ds_store_b64
      const float4 vv = *(const float4*)(V + gbase);
      sVt[(c4 * 4 + 0) * BK + row] = (__bf16)vv.x;
      sVt[(c4 * 4 + 1) * BK + row] = (__bf16)vv.y;
      sVt[(c4 * 4 + 2) * BK + row] = (__bf16)vv.z;
      sVt[(c4 * 4 + 3) * BK + row] = (__bf16)vv.w;
    }

    // Prefetch next tile (256 threads x 256B cover 64 rows x 1KB) -> global_prefetch_b8
    if (kb + BK < S) {
      const int prow = tid >> 2;
      const int pcol = (tid & 3) * 64;
      const size_t pbase = ((size_t)(b * S + kb + BK + prow)) * D_MODEL + pcol;
      __builtin_prefetch((const char*)(K + pbase), 0, 0);
      __builtin_prefetch((const char*)(V + pbase), 0, 0);
    }

    __syncthreads();  // tile visible to all waves

    // ---- Scores: S(16x64) = Qs @ K^T  (four 16x16 C fragments) ----
    v8f c[4];
#pragma unroll
    for (int j = 0; j < 4; ++j) c[j] = (v8f)(0.0f);
#pragma unroll
    for (int f = 0; f < 8; ++f) {
      const int base = f * 32;
#pragma unroll
      for (int j = 0; j < 4; ++j) {
        const __bf16* krow = &sK[(j * 16 + r) * D_MODEL + base];
        v16bf kbf = frag_from_lds(krow + koff, krow + 16 + koff);
        c[j] = __builtin_amdgcn_wmma_f32_16x16x32_bf16(false, qf[f], false, kbf,
                                                       (short)0, c[j], false, false);
      }
    }

    // ---- Online softmax (base-2 domain; register v == rows v / v+8) ----
    v8f av;  // per-register alpha vector (enables v_pk_mul_f32 rescale)
#pragma unroll
    for (int v = 0; v < 8; ++v) {
      const float s0 = c[0][v], s1 = c[1][v], s2 = c[2][v], s3 = c[3][v];
      float mx = fmaxf(fmaxf(s0, s1), fmaxf(s2, s3));
      mx = fmaxf(mx, __shfl_xor(mx, 1, 16));
      mx = fmaxf(mx, __shfl_xor(mx, 2, 16));
      mx = fmaxf(mx, __shfl_xor(mx, 4, 16));
      mx = fmaxf(mx, __shfl_xor(mx, 8, 16));
      const float mn = fmaxf(m[v], mx);
      const float a  = exp2f(m[v] - mn);
      const float p0 = exp2f(s0 - mn);
      const float p1 = exp2f(s1 - mn);
      const float p2 = exp2f(s2 - mn);
      const float p3 = exp2f(s3 - mn);
      float rs = (p0 + p1) + (p2 + p3);
      rs += __shfl_xor(rs, 1, 16);
      rs += __shfl_xor(rs, 2, 16);
      rs += __shfl_xor(rs, 4, 16);
      rs += __shfl_xor(rs, 8, 16);
      l[v] = l[v] * a + rs;
      m[v] = mn;
      av[v] = a;
      const int row = v + ((lane < 16) ? 0 : 8);
      __bf16* pr = sPw + row * BK;
      pr[r]      = (__bf16)p0;
      pr[r + 16] = (__bf16)p1;
      pr[r + 32] = (__bf16)p2;
      pr[r + 48] = (__bf16)p3;
    }

    // Rescale O accumulators by alpha (vector multiply -> packed f32 muls).
#pragma unroll
    for (int n = 0; n < 16; ++n) o[n] *= av;

    // ---- O += P(16x64) @ V(64x256): 32 WMMAs over 16 d-fragments ----
    const __bf16* prow = sPw + r * BK;
    v16bf pa0 = frag_from_lds(prow + koff,      prow + 16 + koff);
    v16bf pa1 = frag_from_lds(prow + 32 + koff, prow + 48 + koff);
#pragma unroll
    for (int n = 0; n < 16; ++n) {
      const __bf16* vrow = &sVt[(n * 16 + r) * BK];
      v16bf vb0 = frag_from_lds(vrow + koff,      vrow + 16 + koff);
      v16bf vb1 = frag_from_lds(vrow + 32 + koff, vrow + 48 + koff);
      o[n] = __builtin_amdgcn_wmma_f32_16x16x32_bf16(false, pa0, false, vb0,
                                                     (short)0, o[n], false, false);
      o[n] = __builtin_amdgcn_wmma_f32_16x16x32_bf16(false, pa1, false, vb1,
                                                     (short)0, o[n], false, false);
    }
  }

  // ---- Normalize by running sum (vector multiply) and store fp32 output ----
  v8f invv;
#pragma unroll
  for (int v = 0; v < 8; ++v) invv[v] = 1.0f / l[v];
#pragma unroll
  for (int n = 0; n < 16; ++n) o[n] *= invv;
#pragma unroll
  for (int v = 0; v < 8; ++v) {
    const int row = v + ((lane < 16) ? 0 : 8);
    float* op = Out + ((size_t)(b * S + q0 + row)) * D_MODEL + r;
#pragma unroll
    for (int n = 0; n < 16; ++n) op[n * 16] = o[n][v];
  }
}

extern "C" void kernel_launch(void* const* d_in, const int* in_sizes, int n_in,
                              void* d_out, int out_size, void* d_ws, size_t ws_size,
                              hipStream_t stream) {
  const float* Q = (const float*)d_in[0];
  const float* K = (const float*)d_in[1];
  const float* V = (const float*)d_in[2];
  float* Out     = (float*)d_out;

  const int S = SEQ_LEN;
  const int B = in_sizes[0] / (S * D_MODEL);

  dim3 grid(S / (WAVES * BQ), B);
  dim3 block(256);
  hipLaunchKernelGGL(flash_attn_bf16_wmma, grid, block, SMEM_BYTES, stream, Q, K, V, Out, S, B);
}
